// SimpleGAT_88433376624938
// MI455X (gfx1250) — matrix operand order repaired and checked by reading, not compiled, not run on hardware
//
#include <hip/hip_runtime.h>
#include <hip/hip_bf16.h>
#include <hip/hip_fp16.h>

// ---------------------------------------------------------------------------
// Types for CDNA5 WMMA (wave32, v_wmma_f32_16x16x32_f16)
// ---------------------------------------------------------------------------
typedef __attribute__((ext_vector_type(16))) _Float16 half16_t;
typedef __attribute__((ext_vector_type(8)))  _Float16 half8_t;
typedef __attribute__((ext_vector_type(4)))  _Float16 half4_t;
typedef __attribute__((ext_vector_type(8)))  float    floatx8;

#define NEG_SLOPE 0.2f

// ---------------------------------------------------------------------------
// Small helper kernels
// ---------------------------------------------------------------------------
__global__ void k_cast_f16(const float* __restrict__ in, _Float16* __restrict__ out, int n) {
    int i = blockIdx.x * blockDim.x + threadIdx.x;
    if (i < n) out[i] = (_Float16)in[i];
}

// W is K x F row-major (fp32); write Wt = F x K (f16) so WMMA B-fragment loads
// are contiguous 16B vectors per lane.
__global__ void k_transpose_cast(const float* __restrict__ W, _Float16* __restrict__ Wt,
                                 int K, int F) {
    int i = blockIdx.x * blockDim.x + threadIdx.x;
    if (i < K * F) {
        int k = i / F, f = i % F;
        Wt[(size_t)f * K + k] = (_Float16)W[i];
    }
}

// u[k] = sum_f W[k][f] * a[f]   (fuses x@W@a into x@(W a) — kills the hd GEMM)
__global__ void k_proj_vec(const float* __restrict__ W, const float* __restrict__ a,
                           float* __restrict__ u, int K, int F) {
    int k = blockIdx.x * blockDim.x + threadIdx.x;
    if (k < K) {
        float s = 0.f;
        for (int f = 0; f < F; ++f) s += W[(size_t)k * F + f] * a[f];
        u[k] = s;
    }
}

// s_src[n] = x[n]·u ; s_dst[n] = x[n]·v  (shared x loads)
__global__ void k_node_scores(const _Float16* __restrict__ x, const float* __restrict__ u,
                              const float* __restrict__ v, float* __restrict__ ss,
                              float* __restrict__ sd, int N, int K) {
    int n = blockIdx.x * blockDim.x + threadIdx.x;
    if (n < N) {
        const _Float16* xp = x + (size_t)n * K;
        float a = 0.f, b = 0.f;
        for (int k = 0; k < K; ++k) {
            float xv = (float)xp[k];
            a += xv * u[k];
            b += xv * v[k];
        }
        ss[n] = a;
        sd[n] = b;
    }
}

__global__ void k_init_nodes(float* __restrict__ m, float* __restrict__ denom, int N) {
    int i = blockIdx.x * blockDim.x + threadIdx.x;
    if (i < N) { m[i] = -__builtin_inff(); denom[i] = 0.f; }
}

__global__ void k_add_vec(const float* __restrict__ a, const float* __restrict__ b,
                          float* __restrict__ o, int n) {
    int i = blockIdx.x * blockDim.x + threadIdx.x;
    if (i < n) o[i] = a[i] + b[i];
}

// float atomic max via sign-split int trick -> global_atomic_max/min
__device__ __forceinline__ void atomicMaxF(float* addr, float val) {
    if (val >= 0.f) atomicMax((int*)addr, __float_as_int(val));
    else            atomicMin((unsigned int*)addr, __float_as_uint(val));
}

__device__ __forceinline__ float leaky(float v) { return v >= 0.f ? v : NEG_SLOPE * v; }

__global__ void k_edge_max(const int* __restrict__ src, const int* __restrict__ dst,
                           const float* __restrict__ ss, const float* __restrict__ sd,
                           float* __restrict__ m, int E) {
    int e = blockIdx.x * blockDim.x + threadIdx.x;
    if (e < E) {
        float v = leaky(ss[src[e]] + sd[dst[e]]);
        atomicMaxF(&m[dst[e]], v);
    }
}

__global__ void k_edge_sum(const int* __restrict__ src, const int* __restrict__ dst,
                           const float* __restrict__ ss, const float* __restrict__ sd,
                           const float* __restrict__ m, float* __restrict__ denom, int E) {
    int e = blockIdx.x * blockDim.x + threadIdx.x;
    if (e < E) {
        int d = dst[e];
        float v = leaky(ss[src[e]] + sd[d]);
        atomicAdd(&denom[d], __expf(v - m[d]));
    }
}

// One wave per edge: lane 0..31 each stream 4-half chunks of hs[src], fp32
// atomicAdd into agg[dst]. L2-resident atomics at 23.3TB/s-backed bandwidth.
__global__ __launch_bounds__(256)
void k_edge_scatter_wave(const int* __restrict__ src, const int* __restrict__ dst,
                         const float* __restrict__ ss, const float* __restrict__ sd,
                         const float* __restrict__ m, const float* __restrict__ denom,
                         const _Float16* __restrict__ hs, float* __restrict__ agg,
                         int E, int F) {
    int gtid = blockIdx.x * blockDim.x + threadIdx.x;
    int e    = gtid >> 5;
    int lane = threadIdx.x & 31;
    if (e >= E) return;
    int s = src[e], d = dst[e];
    float v     = leaky(ss[s] + sd[d]);
    float alpha = __expf(v - m[d]) / (denom[d] + 1e-16f);
    const _Float16* hp = hs + (size_t)s * F;
    float* ap = agg + (size_t)d * F;
    for (int f = lane * 4; f < F; f += 128) {
        half4_t h4 = *(const half4_t*)(hp + f);
        atomicAdd(ap + f + 0, alpha * (float)h4[0]);
        atomicAdd(ap + f + 1, alpha * (float)h4[1]);
        atomicAdd(ap + f + 2, alpha * (float)h4[2]);
        atomicAdd(ap + f + 3, alpha * (float)h4[3]);
    }
}

// Thread-per-edge scatter for the skinny final layer (F=5)
__global__ void k_edge_scatter_small(const int* __restrict__ src, const int* __restrict__ dst,
                                     const float* __restrict__ ss, const float* __restrict__ sd,
                                     const float* __restrict__ m, const float* __restrict__ denom,
                                     const float* __restrict__ hs, float* __restrict__ out,
                                     int E, int F) {
    int e = blockIdx.x * blockDim.x + threadIdx.x;
    if (e >= E) return;
    int s = src[e], d = dst[e];
    float v     = leaky(ss[s] + sd[d]);
    float alpha = __expf(v - m[d]) / (denom[d] + 1e-16f);
    for (int j = 0; j < F; ++j)
        atomicAdd(&out[(size_t)d * F + j], alpha * hs[(size_t)s * F + j]);
}

__global__ void k_relu_cast(const float* __restrict__ agg, _Float16* __restrict__ h, size_t n) {
    size_t i = (size_t)blockIdx.x * blockDim.x + threadIdx.x;
    if (i < n) h[i] = (_Float16)fmaxf(agg[i], 0.f);
}

// Plain dot-product GEMM for the F=5 output layer (WMMA would waste 11/16 cols)
__global__ void k_small_gemm(const _Float16* __restrict__ X, const float* __restrict__ W,
                             const float* __restrict__ bias, float* __restrict__ out,
                             int N, int K, int F) {
    int i = blockIdx.x * blockDim.x + threadIdx.x;
    if (i >= N * F) return;
    int n = i / F, j = i % F;
    float s = bias ? bias[j] : 0.f;
    const _Float16* xp = X + (size_t)n * K;
    for (int k = 0; k < K; ++k) s += (float)xp[k] * W[(size_t)k * F + j];
    out[i] = s;
}

// ---------------------------------------------------------------------------
// WMMA GEMM: C[N x F] = A[N x K](f16) * Bt^T  (Bt stored F x K, f16)
// Block = 256 threads = 8 waves; each wave owns one 16x16 output tile,
// the block covers 16 rows x 128 cols. A strip staged in LDS (+8-half pad
// => 4-bank row skew, conflict-free ds_load_b128). K-loop in steps of 32
// issuing v_wmma_f32_16x16x32_f16 with fp32 accumulation.
// Requires: N%16==0, F%128==0, K%32==0 (true here: 20000 / 512|256 / 256|512).
// ---------------------------------------------------------------------------
__global__ __launch_bounds__(256)
void k_gemm_wmma(const _Float16* __restrict__ A, const _Float16* __restrict__ Bt,
                 const float* __restrict__ bias, float* __restrict__ Cf,
                 _Float16* __restrict__ Ch, int N, int K, int F) {
    __shared__ _Float16 As[16][520];   // up to K=512 + 8-half pad (16B-aligned rows)

    const int wave = threadIdx.x >> 5;
    const int lane = threadIdx.x & 31;
    const int row0 = blockIdx.y * 16;
    const int col0 = blockIdx.x * 128 + wave * 16;

    // Cooperative A-strip load: 16 rows x K halves, 16B vector chunks
    const int chunks = 2 * K;                // (16*K)/8
    const int kc     = K / 8;
    for (int idx = threadIdx.x; idx < chunks; idx += 256) {
        int r = idx / kc;
        int c = (idx % kc) * 8;
        *(half8_t*)&As[r][c] = *(const half8_t*)&A[(size_t)(row0 + r) * K + c];
    }
    __syncthreads();

    floatx8 acc = {};
    const int r    = lane & 15;
    const int aoff = (lane < 16) ? 0 : 8;    // A layout: lane<16 holds K 0-7/16-23
    const int boff = (lane < 16) ? 0 : 8;    // B layout: lane<16 holds K 0-7/16-23
    const _Float16* bcol = Bt + (size_t)(col0 + r) * K;

    for (int k = 0; k < K; k += 32) {
        half8_t a_lo = *(const half8_t*)&As[r][k + aoff];
        half8_t a_hi = *(const half8_t*)&As[r][k + aoff + 16];
        half8_t b_lo = *(const half8_t*)(bcol + k + boff);
        half8_t b_hi = *(const half8_t*)(bcol + k + boff + 16);
        half16_t av = __builtin_shufflevector(a_lo, a_hi, 0,1,2,3,4,5,6,7,8,9,10,11,12,13,14,15);
        half16_t bv = __builtin_shufflevector(b_lo, b_hi, 0,1,2,3,4,5,6,7,8,9,10,11,12,13,14,15);
        acc = __builtin_amdgcn_wmma_f32_16x16x32_f16(
            /*neg_a=*/false, av, /*neg_b=*/false, bv,
            /*c_mod=*/(short)0, acc, /*reuse_a=*/false, /*reuse_b=*/false);
    }

    // C layout: lane<16 -> col=lane, VGPR r = row r; lane>=16 -> col=lane-16, row 8+r
    const int col   = col0 + r;
    const int mbase = (lane < 16) ? 0 : 8;
    const float bv  = bias ? bias[col] : 0.f;
    for (int mm = 0; mm < 8; ++mm) {
        size_t o = (size_t)(row0 + mbase + mm) * F + col;
        float val = acc[mm] + bv;
        if (Cf) Cf[o] = val;
        if (Ch) Ch[o] = (_Float16)val;
    }
}

// ---------------------------------------------------------------------------
// Host orchestration
// ---------------------------------------------------------------------------
extern "C" void kernel_launch(void* const* d_in, const int* in_sizes, int n_in,
                              void* d_out, int out_size, void* d_ws, size_t ws_size,
                              hipStream_t stream) {
    constexpr int N = 20000, E = 160000, D = 256, H1 = 512, H2 = 256, OUTF = 5;
    constexpr int TPB = 256;

    const float* x   = (const float*)d_in[0];
    const int*   ei  = (const int*)d_in[1];
    const int*   src = ei;
    const int*   dst = ei + E;
    const float* c1Ws = (const float*)d_in[2];
    const float* c1Wd = (const float*)d_in[3];
    const float* c1as = (const float*)d_in[4];
    const float* c1ad = (const float*)d_in[5];
    const float* c1b  = (const float*)d_in[6];
    const float* l1W  = (const float*)d_in[7];
    const float* l1b  = (const float*)d_in[8];
    const float* c2Ws = (const float*)d_in[9];
    const float* c2Wd = (const float*)d_in[10];
    const float* c2as = (const float*)d_in[11];
    const float* c2ad = (const float*)d_in[12];
    const float* c2b  = (const float*)d_in[13];
    const float* l2W  = (const float*)d_in[14];
    const float* l2b  = (const float*)d_in[15];
    const float* c3Ws = (const float*)d_in[16];
    const float* c3Wd = (const float*)d_in[17];
    const float* c3as = (const float*)d_in[18];
    const float* c3ad = (const float*)d_in[19];
    const float* c3b  = (const float*)d_in[20];
    const float* l3W  = (const float*)d_in[21];
    const float* l3b  = (const float*)d_in[22];
    float* out = (float*)d_out;

    // Carve workspace (everything written before read; poison-safe)
    char* p = (char*)d_ws;
    auto carve = [&](size_t bytes) -> void* {
        void* r = (void*)p;
        p += (bytes + 255) & ~(size_t)255;
        return r;
    };
    _Float16* hA    = (_Float16*)carve((size_t)N * H1 * sizeof(_Float16)); // ping
    _Float16* hB    = (_Float16*)carve((size_t)N * H1 * sizeof(_Float16)); // pong
    _Float16* hs16  = (_Float16*)carve((size_t)N * H1 * sizeof(_Float16));
    float*    agg   = (float*)carve((size_t)N * H1 * sizeof(float));
    float*    ssrc  = (float*)carve((size_t)N * sizeof(float));
    float*    sdst  = (float*)carve((size_t)N * sizeof(float));
    float*    mbuf  = (float*)carve((size_t)N * sizeof(float));
    float*    denom = (float*)carve((size_t)N * sizeof(float));
    float*    uvec  = (float*)carve(512 * sizeof(float));
    float*    vvec  = (float*)carve(512 * sizeof(float));
    float*    biasc = (float*)carve(512 * sizeof(float));
    _Float16* Wt    = (_Float16*)carve((size_t)512 * 256 * sizeof(_Float16)); // reused
    float*    hs3   = (float*)carve((size_t)N * OUTF * sizeof(float));

    auto gb = [](long long n, int t) { return (int)((n + t - 1) / t); };

    // --- attention softmax scaffolding shared by all layers -----------------
    auto attn_prep = [&](const _Float16* xin, int K, int F,
                         const float* Ws, const float* Wd,
                         const float* as_, const float* ad_) {
        k_proj_vec<<<gb(K, TPB), TPB, 0, stream>>>(Ws, as_, uvec, K, F);
        k_proj_vec<<<gb(K, TPB), TPB, 0, stream>>>(Wd, ad_, vvec, K, F);
        k_node_scores<<<gb(N, TPB), TPB, 0, stream>>>(xin, uvec, vvec, ssrc, sdst, N, K);
        k_init_nodes<<<gb(N, TPB), TPB, 0, stream>>>(mbuf, denom, N);
        k_edge_max<<<gb(E, TPB), TPB, 0, stream>>>(src, dst, ssrc, sdst, mbuf, E);
        k_edge_sum<<<gb(E, TPB), TPB, 0, stream>>>(src, dst, ssrc, sdst, mbuf, denom, E);
    };

    // --- full WMMA layer (layers 1 & 2) -------------------------------------
    auto run_layer = [&](const _Float16* xin, int K, int F,
                         const float* Ws, const float* Wd,
                         const float* as_, const float* ad_, const float* bconv,
                         const float* Wlin, const float* blin, _Float16* hout) {
        attn_prep(xin, K, F, Ws, Wd, as_, ad_);

        dim3 grid(F / 128, N / 16);
        // hs = x @ Ws  (f16 out, consumed by scatter)
        k_transpose_cast<<<gb((long long)K * F, TPB), TPB, 0, stream>>>(Ws, Wt, K, F);
        k_gemm_wmma<<<grid, TPB, 0, stream>>>(xin, Wt, nullptr, nullptr, hs16, N, K, F);
        // agg = x @ Wlin + (blin + bconv)   (fp32, becomes the accumulator)
        k_add_vec<<<gb(F, TPB), TPB, 0, stream>>>(bconv, blin, biasc, F);
        k_transpose_cast<<<gb((long long)K * F, TPB), TPB, 0, stream>>>(Wlin, Wt, K, F);
        k_gemm_wmma<<<grid, TPB, 0, stream>>>(xin, Wt, biasc, agg, nullptr, N, K, F);
        // agg += sum_e alpha_e * hs[src_e]
        k_edge_scatter_wave<<<gb((long long)E * 32, TPB), TPB, 0, stream>>>(
            src, dst, ssrc, sdst, mbuf, denom, hs16, agg, E, F);
        // h_next = relu(agg), cast to f16 for the next layer's WMMA
        k_relu_cast<<<gb((long long)N * F, TPB), TPB, 0, stream>>>(agg, hout, (size_t)N * F);
    };

    // ---- pipeline -----------------------------------------------------------
    k_cast_f16<<<gb((long long)N * D, TPB), TPB, 0, stream>>>(x, hA, N * D);

    run_layer(hA, D,  H1, c1Ws, c1Wd, c1as, c1ad, c1b, l1W, l1b, hB); // 256 -> 512
    run_layer(hB, H1, H2, c2Ws, c2Wd, c2as, c2ad, c2b, l2W, l2b, hA); // 512 -> 256

    // ---- layer 3 (F=5, skinny: plain dot kernels) ---------------------------
    attn_prep(hA, H2, OUTF, c3Ws, c3Wd, c3as, c3ad);
    k_small_gemm<<<gb((long long)N * OUTF, TPB), TPB, 0, stream>>>(
        hA, c3Ws, nullptr, hs3, N, H2, OUTF);
    k_add_vec<<<gb(OUTF, TPB), TPB, 0, stream>>>(c3b, l3b, biasc, OUTF);
    k_small_gemm<<<gb((long long)N * OUTF, TPB), TPB, 0, stream>>>(
        hA, l3W, biasc, out, N, H2, OUTF);
    k_edge_scatter_small<<<gb(E, TPB), TPB, 0, stream>>>(
        src, dst, ssrc, sdst, mbuf, denom, hs3, out, E, OUTF);
}
